// FeatureMatching_55336358641936
// MI455X (gfx1250) — matrix operand order repaired and checked by reading, not compiled, not run on hardware
//
#include <hip/hip_runtime.h>
#include <math.h>

// ---------------------------------------------------------------------------
// FLOT-style scene flow network for MI455X (gfx1250, wave32, WMMA).
// Dense GEMMs (set-abstraction MLPs, feature correlation, fc1) run on
// v_wmma_f32_16x16x32_f16 with f32 accumulation:
//   - weights pre-transposed to [N,K] f16, K padded to a multiple of 32 so a
//     lane's fragment is two contiguous 16-byte loads (CDNA5 16-bit layouts)
//   - the block-shared B tile is staged into LDS with the gfx1250 async copy
//     (global_load_async_to_lds_b128 + s_wait_asynccnt), WMMA B fragments are
//     then read from LDS (ds_load_b128), A streams straight from global/L2
//   - each wave computes NT (2 or 4) adjacent 16x16 tiles: one A fragment is
//     reused across NT WMMAs for 2-4x math per byte loaded
// Instance-norm cancels conv biases exactly, so they are skipped.
// ---------------------------------------------------------------------------

typedef _Float16 h8  __attribute__((ext_vector_type(8)));
typedef _Float16 h16 __attribute__((ext_vector_type(16)));
typedef float    f8  __attribute__((ext_vector_type(8)));

#define CDIVL(a, b) (((a) + (b) - 1) / (b))

// ---------------- input index map (setup_inputs flattening order) ----------
enum {
  IN_PC1S1 = 0, IN_PC1S4, IN_PC2S4, IN_F1, IN_F2,
  W_G1_0 = 5,   B_G1_0, W_G1_1, B_G1_1, W_G1_2, B_G1_2,
  W_G2_0 = 11,  B_G2_0, W_G2_1, B_G2_1, W_G2_2, B_G2_2,
  W_G3_0 = 17,  B_G3_0, W_G3_1, B_G3_1, W_G3_2, B_G3_2,
  W_FM1_0 = 23, B_FM1_0, W_FM1_1, B_FM1_1, W_FM1_2, B_FM1_2,
  W_FM2_0 = 29, B_FM2_0, W_FM2_1, B_FM2_1, W_FM2_2, B_FM2_2,
  W_FR1_0 = 35, B_FR1_0, W_FR1_1, B_FR1_1,
  W_FR2_0 = 39, B_FR2_0, W_FR2_1, B_FR2_1,
  W_FC1 = 43,   B_FC1,   W_FC2 = 45, B_FC2,
  P_GAMMA = 47, P_EPS = 48
};

// ---------------- [B,C,N] -> [B,N,C] ---------------------------------------
__global__ void k_transpose(const float* __restrict__ in, float* __restrict__ out,
                            int Bn, int C, int N) {
  long gid = (long)blockIdx.x * blockDim.x + threadIdx.x;
  long tot = (long)Bn * C * N;
  if (gid >= tot) return;
  int n = (int)(gid % N);
  int c = (int)((gid / N) % C);
  int b = (int)(gid / ((long)C * N));
  out[((long)b * N + n) * C + c] = in[gid];
}

// ---------------- weights f32 [Cin,Cout] -> f16 transposed [Cout,Cpad] -----
__global__ void k_wprep(const float* __restrict__ W, _Float16* __restrict__ wt,
                        int Cin, int Cout, int Cpad) {
  long gid = (long)blockIdx.x * blockDim.x + threadIdx.x;
  if (gid >= (long)Cout * Cpad) return;
  int k = (int)(gid % Cpad);
  int n = (int)(gid / Cpad);
  wt[gid] = (k < Cin) ? (_Float16)W[(long)k * Cout + n] : (_Float16)0.f;
}

// ---------------- generic f32 rows -> f16 padded rows -----------------------
__global__ void k_rows_half(const float* __restrict__ h, int C,
                            _Float16* __restrict__ A, int Cpad, long M) {
  long gid = (long)blockIdx.x * blockDim.x + threadIdx.x;
  if (gid >= M * Cpad) return;
  long row = gid / Cpad;
  int c = (int)(gid % Cpad);
  A[gid] = (c < C) ? (_Float16)h[row * C + c] : (_Float16)0.f;
}

// ---------------- farthest point sampling (one block per batch) -------------
__global__ void k_fps(const float* __restrict__ xyz, int N, int S,
                      float* __restrict__ dist, int* __restrict__ outIdx) {
  int b = blockIdx.x;
  const float* P = xyz + (long)b * N * 3;
  float* D = dist + (long)b * N;
  int* O = outIdx + (long)b * S;
  int t = threadIdx.x, nt = blockDim.x;
  __shared__ float sval[256];
  __shared__ int   sidx[256];
  __shared__ int   sfar;
  for (int i = t; i < N; i += nt) D[i] = 1e10f;
  if (t == 0) sfar = 0;
  __syncthreads();
  for (int it = 0; it < S; ++it) {
    int far = sfar;
    if (t == 0) O[it] = far;
    float cx = P[far * 3], cy = P[far * 3 + 1], cz = P[far * 3 + 2];
    float bv = -1.f; int bi = 0;
    for (int i = t; i < N; i += nt) {
      float dx = P[i * 3] - cx, dy = P[i * 3 + 1] - cy, dz = P[i * 3 + 2] - cz;
      float dm = fminf(D[i], dx * dx + dy * dy + dz * dz);
      D[i] = dm;
      if (dm > bv) { bv = dm; bi = i; }
    }
    sval[t] = bv; sidx[t] = bi;
    __syncthreads();
    for (int s = nt >> 1; s > 0; s >>= 1) {
      if (t < s && sval[t + s] > sval[t]) { sval[t] = sval[t + s]; sidx[t] = sidx[t + s]; }
      __syncthreads();
    }
    if (t == 0) sfar = sidx[0];
    __syncthreads();
  }
}

// ---------------- gather 3-vectors by index --------------------------------
__global__ void k_gather3(const float* __restrict__ xyz, const int* __restrict__ idx,
                          int N, int S, float* __restrict__ out, int Bn) {
  long gid = (long)blockIdx.x * blockDim.x + threadIdx.x;
  if (gid >= (long)Bn * S) return;
  int b = (int)(gid / S);
  int j = idx[gid];
  const float* P = xyz + ((long)b * N + j) * 3;
  out[gid * 3 + 0] = P[0]; out[gid * 3 + 1] = P[1]; out[gid * 3 + 2] = P[2];
}

// ---------------- brute-force kNN (thread per query, insertion top-k) -------
__global__ void k_knn(const float* __restrict__ q, const float* __restrict__ p,
                      int S, int N, int Ks, int* __restrict__ out, int Bn) {
  long gid = (long)blockIdx.x * blockDim.x + threadIdx.x;
  if (gid >= (long)Bn * S) return;
  int b = (int)(gid / S);
  const float* Q = q + gid * 3;
  const float* P = p + (long)b * N * 3;
  float qx = Q[0], qy = Q[1], qz = Q[2];
  float bd[32]; int bi[32];
  for (int i = 0; i < Ks; ++i) { bd[i] = 3.4e38f; bi[i] = 0; }
  for (int j = 0; j < N; ++j) {
    float dx = P[j * 3] - qx, dy = P[j * 3 + 1] - qy, dz = P[j * 3 + 2] - qz;
    float d = dx * dx + dy * dy + dz * dz;
    if (d < bd[Ks - 1]) {
      int pos = Ks - 1;
      while (pos > 0 && bd[pos - 1] > d) {
        bd[pos] = bd[pos - 1]; bi[pos] = bi[pos - 1]; --pos;
      }
      bd[pos] = d; bi[pos] = j;
    }
  }
  int* O = out + gid * Ks;
  for (int i = 0; i < Ks; ++i) O[i] = bi[i];
}

// ---------------- build grouped A matrix: [centered xyz | feats | pad] ------
__global__ void k_group(const float* __restrict__ xyz, const float* __restrict__ feats,
                        int N, int Cf, const float* __restrict__ nx,
                        const int* __restrict__ knn, int S, int Ks,
                        _Float16* __restrict__ A, int Cpad, int Bn) {
  long gid = (long)blockIdx.x * blockDim.x + threadIdx.x;
  long M = (long)Bn * S * Ks;
  if (gid >= M) return;
  int b = (int)(gid / ((long)S * Ks));
  long r = gid % ((long)S * Ks);
  int s = (int)(r / Ks);
  int kk = (int)(r % Ks);
  int j = knn[((long)b * S + s) * Ks + kk];
  const float* P = xyz + ((long)b * N + j) * 3;
  const float* C = nx + ((long)b * S + s) * 3;
  _Float16* row = A + gid * Cpad;
  row[0] = (_Float16)(P[0] - C[0]);
  row[1] = (_Float16)(P[1] - C[1]);
  row[2] = (_Float16)(P[2] - C[2]);
  const float* F = feats + ((long)b * N + j) * Cf;
  for (int c = 0; c < Cf; ++c) row[3 + c] = (_Float16)F[c];
  for (int c = 3 + Cf; c < Cpad; ++c) row[c] = (_Float16)0.f;
}

// ---------------- WMMA f16 GEMM: C[M,N] = A[M,K] * Bt[N,K]^T ---------------
// Block = 4 waves (128 threads), each wave owns one 16-row M tile and NT
// adjacent 16-col N tiles. The block's shared B tile (NT*16 rows x K) is
// staged into LDS via gfx1250 async copy, then each K-step does:
//   2x global b128 (A frag) + 2*NT ds b128 (B frags) + NT wmma.
// K must be a multiple of 32 and <= 256. Fragment index math follows the
// CDNA5 ISA 16-bit A (16x32) / B (32x16) lane layouts.
#define LDS_KP 264   // 256 + 8-halves row pad (16B) to spread LDS banks

template <int NT>
__global__ void __launch_bounds__(128)
k_gemm_f16_wmma(const _Float16* __restrict__ A, long strA,
                const _Float16* __restrict__ B, long strB,
                float* __restrict__ C, long strC,
                int M, int N, int K) {
  __shared__ _Float16 lB[64 * LDS_KP];
  const int lane = threadIdx.x & 31;
  const int wave = threadIdx.x >> 5;
  const int tm = blockIdx.x * 4 + wave;
  const int tn = blockIdx.y;
  A += (long)blockIdx.z * strA;
  B += (long)blockIdx.z * strB;
  C += (long)blockIdx.z * strC;

  // ---- async-stage the B tile (rows tn*NT*16 .. +NT*16-1, all K) into LDS
  {
    const _Float16* bsrc = B + (long)tn * (NT * 16) * K;
    const int ck = K / 8;              // 16-byte chunks per row
    const int nchunk = NT * 16 * ck;
    for (int t = threadIdx.x; t < nchunk; t += 128) {
      int row = t / ck, c8 = t % ck;
      unsigned loff = (unsigned)(unsigned long)(&lB[row * LDS_KP + c8 * 8]);
      const _Float16* g = bsrc + (long)row * K + c8 * 8;
      asm volatile("global_load_async_to_lds_b128 %0, %1, off"
                   :: "v"(loff), "v"(g) : "memory");
    }
    asm volatile("s_wait_asynccnt 0x0" ::: "memory");
  }
  __syncthreads();

  if (tm * 16 < M) {                   // uniform per wave: EXEC stays full
    const int l15 = lane & 15;
    const int hi = lane >> 4;
    const _Float16* ar = A + (long)(tm * 16 + l15) * K + (hi << 3);  // k-sub 0/8
    f8 acc[NT];
#pragma unroll
    for (int i = 0; i < NT; ++i)
      acc[i] = (f8){0.f, 0.f, 0.f, 0.f, 0.f, 0.f, 0.f, 0.f};
    for (int k0 = 0; k0 < K; k0 += 32) {
      h8 a0 = *(const h8*)(ar + k0);
      h8 a1 = *(const h8*)(ar + k0 + 16);
      if (k0 + 32 < K) __builtin_prefetch(ar + k0 + 32, 0, 1);
      h16 av = __builtin_shufflevector(a0, a1, 0, 1, 2, 3, 4, 5, 6, 7,
                                       8, 9, 10, 11, 12, 13, 14, 15);
#pragma unroll
      for (int nt = 0; nt < NT; ++nt) {
        const _Float16* bp = &lB[(nt * 16 + l15) * LDS_KP + k0 + (hi << 3)];
        h8 b0 = *(const h8*)(bp);
        h8 b1 = *(const h8*)(bp + 16);
        h16 bv = __builtin_shufflevector(b0, b1, 0, 1, 2, 3, 4, 5, 6, 7,
                                         8, 9, 10, 11, 12, 13, 14, 15);
        acc[nt] = __builtin_amdgcn_wmma_f32_16x16x32_f16(
            false, av, false, bv, (short)0, acc[nt], false, false);
      }
    }
#pragma unroll
    for (int nt = 0; nt < NT; ++nt) {
      float* crow = C + (long)(tm * 16 + (hi << 3)) * N + tn * (NT * 16) + nt * 16 + l15;
#pragma unroll
      for (int r = 0; r < 8; ++r) crow[(long)r * N] = acc[nt][r];
    }
  }
}

// ---------------- instance norm: per (batch, channel) stats ----------------
__global__ void k_in_stats(const float* __restrict__ h, int SK, int C,
                           float* __restrict__ mu, float* __restrict__ rv) {
  int bc = blockIdx.x;
  int b = bc / C, c = bc % C;
  const float* H = h + (long)b * SK * C + c;
  float s = 0.f, s2 = 0.f;
  for (int i = threadIdx.x; i < SK; i += blockDim.x) {
    float v = H[(long)i * C];
    s += v; s2 += v * v;
  }
  __shared__ float sh[256], sh2[256];
  sh[threadIdx.x] = s; sh2[threadIdx.x] = s2;
  __syncthreads();
  for (int st = 128; st > 0; st >>= 1) {
    if (threadIdx.x < st) { sh[threadIdx.x] += sh[threadIdx.x + st];
                            sh2[threadIdx.x] += sh2[threadIdx.x + st]; }
    __syncthreads();
  }
  if (threadIdx.x == 0) {
    float m = sh[0] / (float)SK;
    float v = sh2[0] / (float)SK - m * m;
    mu[bc] = m;
    rv[bc] = rsqrtf(fmaxf(v, 0.f) + 1e-5f);
  }
}

__global__ void k_in_apply(float* __restrict__ h, int SK, int C,
                           const float* __restrict__ mu, const float* __restrict__ rv,
                           long total) {
  long gid = (long)blockIdx.x * blockDim.x + threadIdx.x;
  if (gid >= total) return;
  long row = gid / C;
  int c = (int)(gid % C);
  int b = (int)(row / SK);
  float v = (h[gid] - mu[b * C + c]) * rv[b * C + c];
  h[gid] = v > 0.f ? v : 0.f;
}

// ---------------- max over the K-sample dimension --------------------------
__global__ void k_maxpool(const float* __restrict__ h, int S, int Ks, int C,
                          float* __restrict__ out, int Bn) {
  long gid = (long)blockIdx.x * blockDim.x + threadIdx.x;
  long tot = (long)Bn * S * C;
  if (gid >= tot) return;
  int c = (int)(gid % C);
  int s = (int)((gid / C) % S);
  int b = (int)(gid / ((long)C * S));
  long base = ((long)(b * S + s) * Ks) * C + c;
  float m = -3.4e38f;
  for (int k = 0; k < Ks; ++k) m = fmaxf(m, h[base + (long)k * C]);
  out[((long)b * S + s) * C + c] = m;
}

// ---------------- L2-normalize feature rows -> f16 -------------------------
__global__ void k_featnorm(const float* __restrict__ f, _Float16* __restrict__ o,
                           int SC, int C, int Bn) {
  long gid = (long)blockIdx.x * blockDim.x + threadIdx.x;
  if (gid >= (long)Bn * SC) return;
  const float* F = f + gid * C;
  float s = 0.f;
  for (int c = 0; c < C; ++c) s += F[c] * F[c];
  float inv = rsqrtf(s + 1e-6f);
  _Float16* O = o + gid * C;
  for (int c = 0; c < C; ++c) O[c] = (_Float16)(F[c] * inv);
}

// ---------------- K = exp(-(1-corr)/eps) * (|p1-p2|^2 < 1) -----------------
__global__ void k_build_K(const float* __restrict__ corr,
                          const float* __restrict__ p1, const float* __restrict__ p2,
                          const float* __restrict__ epsp,
                          float* __restrict__ Ko, int Np, int Bn) {
  long gid = (long)blockIdx.x * blockDim.x + threadIdx.x;
  long tot = (long)Bn * Np * Np;
  if (gid >= tot) return;
  int m = (int)(gid % Np);
  int n = (int)((gid / Np) % Np);
  int b = (int)(gid / ((long)Np * Np));
  const float* P1 = p1 + ((long)b * Np + n) * 3;
  const float* P2 = p2 + ((long)b * Np + m) * 3;
  float dx = P1[0] - P2[0], dy = P1[1] - P2[1], dz = P1[2] - P2[2];
  float dm = dx * dx + dy * dy + dz * dz;
  float eps = expf(epsp[0]) + 0.03f;
  Ko[gid] = (dm < 1.f) ? expf(-(1.f - corr[gid]) / eps) : 0.f;
}

__global__ void k_init_ab(float* __restrict__ a, float* __restrict__ bvec,
                          int Np, int Bn) {
  long gid = (long)blockIdx.x * blockDim.x + threadIdx.x;
  if (gid >= (long)Bn * Np) return;
  a[gid] = 1.f / (float)Np;
  bvec[gid] = 1.f / (float)Np;
}

// b = (prob2 / (K^T a + 1e-8))^power
__global__ void k_kta(const float* __restrict__ K, const float* __restrict__ a,
                      float* __restrict__ bvec, const float* __restrict__ gam,
                      const float* __restrict__ epsp, int Np, int Bn) {
  long gid = (long)blockIdx.x * blockDim.x + threadIdx.x;
  if (gid >= (long)Bn * Np) return;
  int m = (int)(gid % Np);
  int b = (int)(gid / Np);
  float s = 0.f;
  const float* Kb = K + (long)b * Np * Np;
  const float* ab = a + (long)b * Np;
  for (int n = 0; n < Np; ++n) s += Kb[(long)n * Np + m] * ab[n];
  float g = expf(gam[0]) + 0.03f;
  float e = expf(epsp[0]) + 0.03f;
  float pw = g / (g + e);
  bvec[gid] = powf((1.f / (float)Np) / (s + 1e-8f), pw);
}

// a = (prob1 / (K b + 1e-8))^power
__global__ void k_kb(const float* __restrict__ K, const float* __restrict__ bvec,
                     float* __restrict__ a, const float* __restrict__ gam,
                     const float* __restrict__ epsp, int Np, int Bn) {
  long gid = (long)blockIdx.x * blockDim.x + threadIdx.x;
  if (gid >= (long)Bn * Np) return;
  int n = (int)(gid % Np);
  int b = (int)(gid / Np);
  float s = 0.f;
  const float* Kr = K + ((long)b * Np + n) * Np;
  const float* bb = bvec + (long)b * Np;
  for (int m = 0; m < Np; ++m) s += Kr[m] * bb[m];
  float g = expf(gam[0]) + 0.03f;
  float e = expf(epsp[0]) + 0.03f;
  float pw = g / (g + e);
  a[gid] = powf((1.f / (float)Np) / (s + 1e-8f), pw);
}

// flow_init = (T p2)/(rowsum(T)+1e-6) - p1 with T = a K b^T
__global__ void k_flow_init(const float* __restrict__ K, const float* __restrict__ a,
                            const float* __restrict__ bvec,
                            const float* __restrict__ p2, const float* __restrict__ p1,
                            float* __restrict__ flow, int Np, int Bn) {
  long gid = (long)blockIdx.x * blockDim.x + threadIdx.x;
  if (gid >= (long)Bn * Np) return;
  int n = (int)(gid % Np);
  int b = (int)(gid / Np);
  const float* Kr = K + ((long)b * Np + n) * Np;
  const float* bb = bvec + (long)b * Np;
  const float* P2 = p2 + (long)b * Np * 3;
  float an = a[gid];
  float rs = 0.f, f0 = 0.f, f1 = 0.f, f2 = 0.f;
  for (int m = 0; m < Np; ++m) {
    float T = an * Kr[m] * bb[m];
    rs += T;
    f0 += T * P2[m * 3 + 0];
    f1 += T * P2[m * 3 + 1];
    f2 += T * P2[m * 3 + 2];
  }
  float inv = 1.f / (rs + 1e-6f);
  const float* P1 = p1 + gid * 3;
  flow[gid * 3 + 0] = f0 * inv - P1[0];
  flow[gid * 3 + 1] = f1 * inv - P1[1];
  flow[gid * 3 + 2] = f2 * inv - P1[2];
}

// ---------------- 3-NN inverse-distance feature propagation ----------------
__global__ void k_prop3(const float* __restrict__ x1, int N1,
                        const float* __restrict__ x2, int S2,
                        const float* __restrict__ p2, int C,
                        float* __restrict__ out, int Bn) {
  long gid = (long)blockIdx.x * blockDim.x + threadIdx.x;
  if (gid >= (long)Bn * N1) return;
  int b = (int)(gid / N1);
  const float* Q = x1 + gid * 3;
  const float* P = x2 + (long)b * S2 * 3;
  float qx = Q[0], qy = Q[1], qz = Q[2];
  float d0 = 3.4e38f, d1 = 3.4e38f, d2 = 3.4e38f;
  int i0 = 0, i1 = 0, i2 = 0;
  for (int j = 0; j < S2; ++j) {
    float dx = P[j * 3] - qx, dy = P[j * 3 + 1] - qy, dz = P[j * 3 + 2] - qz;
    float d = dx * dx + dy * dy + dz * dz;
    if (d < d0)      { d2 = d1; i2 = i1; d1 = d0; i1 = i0; d0 = d; i0 = j; }
    else if (d < d1) { d2 = d1; i2 = i1; d1 = d; i1 = j; }
    else if (d < d2) { d2 = d; i2 = j; }
  }
  float w0 = 1.f / (d0 + 1e-8f), w1 = 1.f / (d1 + 1e-8f), w2 = 1.f / (d2 + 1e-8f);
  float ws = w0 + w1 + w2;
  w0 /= ws; w1 /= ws; w2 /= ws;
  const float* F = p2 + (long)b * S2 * C;
  float* O = out + gid * C;
  for (int c = 0; c < C; ++c)
    O[c] = w0 * F[(long)i0 * C + c] + w1 * F[(long)i1 * C + c] + w2 * F[(long)i2 * C + c];
}

// ---------------- fc helpers -----------------------------------------------
__global__ void k_bias_relu(float* __restrict__ h, const float* __restrict__ bias,
                            long M, int C) {
  long gid = (long)blockIdx.x * blockDim.x + threadIdx.x;
  if (gid >= M * C) return;
  float v = h[gid] + bias[gid % C];
  h[gid] = v > 0.f ? v : 0.f;
}

__global__ void k_fc2(const float* __restrict__ h, const float* __restrict__ W2,
                      const float* __restrict__ b2, float* __restrict__ out, long M) {
  long gid = (long)blockIdx.x * blockDim.x + threadIdx.x;
  if (gid >= M) return;
  const float* H = h + gid * 128;
  for (int j = 0; j < 3; ++j) {
    float s = b2[j];
    for (int c = 0; c < 128; ++c) s += H[c] * W2[c * 3 + j];
    out[gid * 3 + j] = s;
  }
}

// ---------------- out[b,c,n] = x[b,n,c] + y[b,n,c] -------------------------
__global__ void k_add_t(const float* __restrict__ x, const float* __restrict__ y,
                        float* __restrict__ o, int N, int Bn) {
  long gid = (long)blockIdx.x * blockDim.x + threadIdx.x;
  long tot = (long)Bn * N * 3;
  if (gid >= tot) return;
  int b = (int)(gid / ((long)N * 3));
  long r = gid % ((long)N * 3);
  int n = (int)(r / 3);
  int c = (int)(r % 3);
  o[((long)b * 3 + c) * N + n] = x[gid] + y[gid];
}

// ===========================================================================
extern "C" void kernel_launch(void* const* d_in, const int* in_sizes, int n_in,
                              void* d_out, int out_size, void* d_ws, size_t ws_size,
                              hipStream_t stream) {
  (void)in_sizes; (void)n_in; (void)out_size; (void)ws_size;
  const int Bn = 8, N1 = 8192, N4 = 2048;
  const int S8 = 1024, S16 = 512, S32 = 256;

  char* wsb = (char*)d_ws;
  size_t off = 0;
  auto alloc = [&](size_t bytes) -> void* {
    size_t a = (off + 255) & ~(size_t)255;
    off = a + bytes;
    return (void*)(wsb + a);
  };

  // big reusable scratch: grouped A matrix (f16) and GEMM output (f32)
  _Float16* ABUF = (_Float16*)alloc((size_t)88 * 1024 * 1024 * sizeof(_Float16));
  float*    HBUF = (float*)alloc((size_t)68 * 1024 * 1024 * sizeof(float));
  float* FPSD = (float*)alloc((size_t)Bn * N4 * sizeof(float));
  int*   FPSI = (int*)alloc((size_t)Bn * N4 * sizeof(int));
  int*   KNN  = (int*)alloc((size_t)Bn * N4 * 32 * sizeof(int));
  float* MUB  = (float*)alloc((size_t)Bn * 256 * sizeof(float));
  float* RVB  = (float*)alloc((size_t)Bn * 256 * sizeof(float));

  auto gemm = [&](const _Float16* A, long sA, const _Float16* Bt, long sB,
                  float* Cc, long sC, int M, int N, int K, int nb) {
    if (N % 64 == 0) {
      dim3 grid((unsigned)((M + 63) / 64), (unsigned)(N / 64), (unsigned)nb);
      k_gemm_f16_wmma<4><<<grid, dim3(128), 0, stream>>>(A, sA, Bt, sB, Cc, sC, M, N, K);
    } else {  // N == 32
      dim3 grid((unsigned)((M + 63) / 64), (unsigned)(N / 32), (unsigned)nb);
      k_gemm_f16_wmma<2><<<grid, dim3(128), 0, stream>>>(A, sA, Bt, sB, Cc, sC, M, N, K);
    }
  };

  auto prep_w = [&](int widx, int Cin, int Cout, int Cpad) -> _Float16* {
    _Float16* wt = (_Float16*)alloc((size_t)Cout * Cpad * sizeof(_Float16));
    long tot = (long)Cout * Cpad;
    k_wprep<<<dim3((unsigned)CDIVL(tot, 256)), dim3(256), 0, stream>>>(
        (const float*)d_in[widx], wt, Cin, Cout, Cpad);
    return wt;
  };

  // PointNetSetAbstraction (knn grouping, instance-norm(+bias cancels), relu, maxpool)
  auto run_sa = [&](const float* xyz, int N, const float* feats, int Cf,
                    int S, int Ks, const int* widx, const int* couts, int nl,
                    float* newxyz, float* outfeat) {
    k_fps<<<dim3((unsigned)Bn), dim3(256), 0, stream>>>(xyz, N, S, FPSD, FPSI);
    k_gather3<<<dim3((unsigned)CDIVL((long)Bn * S, 256)), dim3(256), 0, stream>>>(
        xyz, FPSI, N, S, newxyz, Bn);
    k_knn<<<dim3((unsigned)CDIVL((long)Bn * S, 128)), dim3(128), 0, stream>>>(
        newxyz, xyz, S, N, Ks, KNN, Bn);
    int Cin = 3 + Cf, Cpad = ((Cin + 31) / 32) * 32;
    long M = (long)Bn * S * Ks;
    k_group<<<dim3((unsigned)CDIVL(M, 256)), dim3(256), 0, stream>>>(
        xyz, feats, N, Cf, newxyz, KNN, S, Ks, ABUF, Cpad, Bn);
    int curC = Cin, curPad = Cpad;
    for (int l = 0; l < nl; ++l) {
      int Co = couts[l];
      _Float16* wt = prep_w(widx[l], curC, Co, curPad);
      gemm(ABUF, 0, wt, 0, HBUF, 0, (int)M, Co, curPad, 1);
      k_in_stats<<<dim3((unsigned)(Bn * Co)), dim3(256), 0, stream>>>(
          HBUF, S * Ks, Co, MUB, RVB);
      long te = M * Co;
      k_in_apply<<<dim3((unsigned)CDIVL(te, 256)), dim3(256), 0, stream>>>(
          HBUF, S * Ks, Co, MUB, RVB, te);
      if (l + 1 < nl) {
        long ta = M * Co;
        k_rows_half<<<dim3((unsigned)CDIVL(ta, 256)), dim3(256), 0, stream>>>(
            HBUF, Co, ABUF, Co, M);
        curC = Co; curPad = Co;
      }
    }
    int Cl = couts[nl - 1];
    long tp = (long)Bn * S * Cl;
    k_maxpool<<<dim3((unsigned)CDIVL(tp, 256)), dim3(256), 0, stream>>>(
        HBUF, S, Ks, Cl, outfeat, Bn);
  };

  // ---- transpose inputs to points-major [B,N,C] ----
  float* P1 = (float*)alloc((size_t)Bn * N1 * 3 * 4);
  float* P4 = (float*)alloc((size_t)Bn * N4 * 3 * 4);
  float* Q4 = (float*)alloc((size_t)Bn * N4 * 3 * 4);
  float* F1 = (float*)alloc((size_t)Bn * N4 * 64 * 4);
  float* F2 = (float*)alloc((size_t)Bn * N4 * 64 * 4);
  k_transpose<<<dim3((unsigned)CDIVL((long)Bn * 3 * N1, 256)), dim3(256), 0, stream>>>(
      (const float*)d_in[IN_PC1S1], P1, Bn, 3, N1);
  k_transpose<<<dim3((unsigned)CDIVL((long)Bn * 3 * N4, 256)), dim3(256), 0, stream>>>(
      (const float*)d_in[IN_PC1S4], P4, Bn, 3, N4);
  k_transpose<<<dim3((unsigned)CDIVL((long)Bn * 3 * N4, 256)), dim3(256), 0, stream>>>(
      (const float*)d_in[IN_PC2S4], Q4, Bn, 3, N4);
  k_transpose<<<dim3((unsigned)CDIVL((long)Bn * 64 * N4, 256)), dim3(256), 0, stream>>>(
      (const float*)d_in[IN_F1], F1, Bn, 64, N4);
  k_transpose<<<dim3((unsigned)CDIVL((long)Bn * 64 * N4, 256)), dim3(256), 0, stream>>>(
      (const float*)d_in[IN_F2], F2, Bn, 64, N4);

  // ---- extract_glob for both clouds ----
  const int wg1[3] = {W_G1_0, W_G1_1, W_G1_2}; const int c128[3] = {128, 128, 128};
  const int wg2[3] = {W_G2_0, W_G2_1, W_G2_2};
  const int wg3[3] = {W_G3_0, W_G3_1, W_G3_2}; const int c256[3] = {256, 256, 256};

  float* x1s8  = (float*)alloc((size_t)Bn * S8 * 3 * 4);
  float* g1f8  = (float*)alloc((size_t)Bn * S8 * 128 * 4);
  float* x1s16 = (float*)alloc((size_t)Bn * S16 * 3 * 4);
  float* g1f16 = (float*)alloc((size_t)Bn * S16 * 128 * 4);
  float* x1s32 = (float*)alloc((size_t)Bn * S32 * 3 * 4);
  float* fg1   = (float*)alloc((size_t)Bn * S32 * 256 * 4);
  run_sa(P4, N4, F1, 64, S8, 32, wg1, c128, 3, x1s8, g1f8);
  run_sa(x1s8, S8, g1f8, 128, S16, 24, wg2, c128, 3, x1s16, g1f16);
  run_sa(x1s16, S16, g1f16, 128, S32, 16, wg3, c256, 3, x1s32, fg1);

  float* x2s8  = (float*)alloc((size_t)Bn * S8 * 3 * 4);
  float* g2f8  = (float*)alloc((size_t)Bn * S8 * 128 * 4);
  float* x2s16 = (float*)alloc((size_t)Bn * S16 * 3 * 4);
  float* g2f16 = (float*)alloc((size_t)Bn * S16 * 128 * 4);
  float* x2s32 = (float*)alloc((size_t)Bn * S32 * 3 * 4);
  float* fg2   = (float*)alloc((size_t)Bn * S32 * 256 * 4);
  run_sa(Q4, N4, F2, 64, S8, 32, wg1, c128, 3, x2s8, g2f8);
  run_sa(x2s8, S8, g2f8, 128, S16, 24, wg2, c128, 3, x2s16, g2f16);
  run_sa(x2s16, S16, g2f16, 128, S32, 16, wg3, c256, 3, x2s32, fg2);

  // ---- flow init: cosine correlation (WMMA batched GEMM) + Sinkhorn ----
  _Float16* F1N = (_Float16*)alloc((size_t)Bn * S32 * 256 * sizeof(_Float16));
  _Float16* F2N = (_Float16*)alloc((size_t)Bn * S32 * 256 * sizeof(_Float16));
  k_featnorm<<<dim3((unsigned)CDIVL((long)Bn * S32, 128)), dim3(128), 0, stream>>>(
      fg1, F1N, S32, 256, Bn);
  k_featnorm<<<dim3((unsigned)CDIVL((long)Bn * S32, 128)), dim3(128), 0, stream>>>(
      fg2, F2N, S32, 256, Bn);
  float* CORR = (float*)alloc((size_t)Bn * S32 * S32 * 4);
  gemm(F1N, (long)S32 * 256, F2N, (long)S32 * 256, CORR, (long)S32 * S32,
       S32, S32, 256, Bn);
  float* KM = (float*)alloc((size_t)Bn * S32 * S32 * 4);
  k_build_K<<<dim3((unsigned)CDIVL((long)Bn * S32 * S32, 256)), dim3(256), 0, stream>>>(
      CORR, x1s32, x2s32, (const float*)d_in[P_EPS], KM, S32, Bn);
  float* AV = (float*)alloc((size_t)Bn * S32 * 4);
  float* BV = (float*)alloc((size_t)Bn * S32 * 4);
  k_init_ab<<<dim3((unsigned)CDIVL((long)Bn * S32, 256)), dim3(256), 0, stream>>>(
      AV, BV, S32, Bn);
  for (int it = 0; it < 10; ++it) {
    k_kta<<<dim3((unsigned)CDIVL((long)Bn * S32, 128)), dim3(128), 0, stream>>>(
        KM, AV, BV, (const float*)d_in[P_GAMMA], (const float*)d_in[P_EPS], S32, Bn);
    k_kb<<<dim3((unsigned)CDIVL((long)Bn * S32, 128)), dim3(128), 0, stream>>>(
        KM, BV, AV, (const float*)d_in[P_GAMMA], (const float*)d_in[P_EPS], S32, Bn);
  }
  float* FLOWI = (float*)alloc((size_t)Bn * S32 * 3 * 4);
  k_flow_init<<<dim3((unsigned)CDIVL((long)Bn * S32, 128)), dim3(128), 0, stream>>>(
      KM, AV, BV, x2s32, x1s32, FLOWI, S32, Bn);

  // ---- upsample_flow ----
  float* tmpx = (float*)alloc((size_t)Bn * N4 * 3 * 4);
  const int wfm1[3] = {W_FM1_0, W_FM1_1, W_FM1_2}; const int cfm1[3] = {32, 32, 64};
  const int wfm2[3] = {W_FM2_0, W_FM2_1, W_FM2_2}; const int cfm2[3] = {64, 64, 128};
  float* FFa = (float*)alloc((size_t)Bn * S16 * 3 * 4);
  k_prop3<<<dim3((unsigned)CDIVL((long)Bn * S16, 128)), dim3(128), 0, stream>>>(
      x1s16, S16, x1s32, S32, FLOWI, 3, FFa, Bn);
  float* FFb = (float*)alloc((size_t)Bn * S16 * 64 * 4);
  run_sa(x1s16, S16, FFa, 3, S16, 16, wfm1, cfm1, 3, tmpx, FFb);
  float* FFc = (float*)alloc((size_t)Bn * S8 * 64 * 4);
  k_prop3<<<dim3((unsigned)CDIVL((long)Bn * S8, 128)), dim3(128), 0, stream>>>(
      x1s8, S8, x1s16, S16, FFb, 64, FFc, Bn);
  float* FFd = (float*)alloc((size_t)Bn * S8 * 128 * 4);
  run_sa(x1s8, S8, FFc, 64, S8, 16, wfm2, cfm2, 3, tmpx, FFd);
  float* FFe = (float*)alloc((size_t)Bn * N4 * 128 * 4);
  k_prop3<<<dim3((unsigned)CDIVL((long)Bn * N4, 128)), dim3(128), 0, stream>>>(
      P4, N4, x1s8, S8, FFd, 128, FFe, Bn);

  // ---- flow regressor ----
  const int wfr1[2] = {W_FR1_0, W_FR1_1};
  const int wfr2[2] = {W_FR2_0, W_FR2_1};
  float* FFf = (float*)alloc((size_t)Bn * N4 * 128 * 4);
  run_sa(P4, N4, FFe, 128, N4, 32, wfr1, c128, 2, tmpx, FFf);
  float* FFg = (float*)alloc((size_t)Bn * N4 * 128 * 4);
  run_sa(P4, N4, FFf, 128, N4, 32, wfr2, c128, 2, tmpx, FFg);

  long Mfc = (long)Bn * N4;  // 16384
  k_rows_half<<<dim3((unsigned)CDIVL(Mfc * 128, 256)), dim3(256), 0, stream>>>(
      FFg, 128, ABUF, 128, Mfc);
  _Float16* wt1 = prep_w(W_FC1, 128, 128, 128);
  gemm(ABUF, 0, wt1, 0, HBUF, 0, (int)Mfc, 128, 128, 1);
  k_bias_relu<<<dim3((unsigned)CDIVL(Mfc * 128, 256)), dim3(256), 0, stream>>>(
      HBUF, (const float*)d_in[B_FC1], Mfc, 128);
  float* FLR = (float*)alloc((size_t)Bn * N4 * 3 * 4);
  k_fc2<<<dim3((unsigned)CDIVL(Mfc, 128)), dim3(128), 0, stream>>>(
      HBUF, (const float*)d_in[W_FC2], (const float*)d_in[B_FC2], FLR, Mfc);

  // ---- final propagation + residual flows ----
  float* FLOW = (float*)alloc((size_t)Bn * N1 * 3 * 4);
  k_prop3<<<dim3((unsigned)CDIVL((long)Bn * N1, 128)), dim3(128), 0, stream>>>(
      P1, N1, P4, N4, FLR, 3, FLOW, Bn);
  float* FLUP = (float*)alloc((size_t)Bn * N1 * 3 * 4);
  k_prop3<<<dim3((unsigned)CDIVL((long)Bn * N1, 128)), dim3(128), 0, stream>>>(
      P1, N1, x1s32, S32, FLOWI, 3, FLUP, Bn);
  float* FLRUP = (float*)alloc((size_t)Bn * N4 * 3 * 4);
  k_prop3<<<dim3((unsigned)CDIVL((long)Bn * N4, 128)), dim3(128), 0, stream>>>(
      P4, N4, x1s32, S32, FLOWI, 3, FLRUP, Bn);

  float* out = (float*)d_out;
  k_add_t<<<dim3((unsigned)CDIVL((long)Bn * N1 * 3, 256)), dim3(256), 0, stream>>>(
      FLOW, FLUP, out, N1, Bn);
  k_add_t<<<dim3((unsigned)CDIVL((long)Bn * N4 * 3, 256)), dim3(256), 0, stream>>>(
      FLR, FLRUP, out + (size_t)Bn * 3 * N1, N4, Bn);
}
